// MultipleCandidateAttention_33689723470274
// MI455X (gfx1250) — compile-verified, matrix operand order checked by reading, not run on hardware
//
#include <hip/hip_runtime.h>
#include <hip/hip_bf16.h>

// CDNA5 (gfx1250) fused additive attention.
// K1/K2: bf16-WMMA projection GEMMs (pf, pq). K3: fused score(tanh)+softmax+
// alpha@feature with the score tile resident in the WGP's 320KB LDS and the
// feature K-slabs staged via async global->LDS (ASYNCcnt) double buffering.

typedef __attribute__((ext_vector_type(16))) __bf16 v16bf;
typedef __attribute__((ext_vector_type(8)))  float  v8f;
typedef __attribute__((ext_vector_type(4)))  int    v4i;

#define WMMA_BF16(a, b, c) \
  __builtin_amdgcn_wmma_f32_16x16x32_bf16(false, (a), false, (b), (short)0, (c), false, false)

#define GLOBAL_AS __attribute__((address_space(1)))
#define LDS_AS    __attribute__((address_space(3)))

constexpr int kB = 16, kF = 2048, kQ = 64, kD = 256, kA = 128;

// ---- native tanh if the toolchain exposes V_TANH_F32 ------------------------
#if __has_builtin(__builtin_amdgcn_tanhf)
__device__ __forceinline__ float fast_tanh(float x) { return __builtin_amdgcn_tanhf(x); }
#elif __has_builtin(__builtin_amdgcn_tanh_f32)
__device__ __forceinline__ float fast_tanh(float x) { return __builtin_amdgcn_tanh_f32(x); }
#else
__device__ __forceinline__ float fast_tanh(float x) {
  const float t = __expf(2.0f * x);                      // v_exp_f32
  return (t - 1.0f) * __builtin_amdgcn_rcpf(t + 1.0f);   // v_rcp_f32, no slow div
}
#endif

// ---- async global->LDS staging (gfx1250 ASYNC path) -------------------------
#if __has_builtin(__builtin_amdgcn_global_load_async_to_lds_b128)
#define HAVE_ASYNC_LDS 1
#else
#define HAVE_ASYNC_LDS 0
#endif

#if HAVE_ASYNC_LDS
template <int N>
__device__ __forceinline__ void wait_async() {
#if __has_builtin(__builtin_amdgcn_s_wait_asynccnt)
  __builtin_amdgcn_s_wait_asynccnt(N);
#else
  asm volatile("s_wait_asynccnt %0" ::"n"(N) : "memory");
#endif
}
// Copy a 32 x kD f32 slab (32KB) global->LDS: 8 x b128 per thread, coalesced.
// Builtin signature (per hipcc diagnostic): (as1 v4i*, as3 v4i*, imm, imm).
__device__ __forceinline__ void async_slab(const float* __restrict__ gsrc,
                                           float* __restrict__ ldst, int tid) {
#pragma unroll
  for (int i = 0; i < 8; ++i) {
    const int idx = (tid + i * 256) * 4;  // float index, 16B aligned
    __builtin_amdgcn_global_load_async_to_lds_b128(
        (GLOBAL_AS v4i*)(gsrc + idx), (LDS_AS v4i*)(ldst + idx), 0, 0);
  }
}
#endif

__device__ __forceinline__ void pack8(v16bf& v, int base, float4 lo, float4 hi) {
  v[base + 0] = (__bf16)lo.x; v[base + 1] = (__bf16)lo.y;
  v[base + 2] = (__bf16)lo.z; v[base + 3] = (__bf16)lo.w;
  v[base + 4] = (__bf16)hi.x; v[base + 5] = (__bf16)hi.y;
  v[base + 6] = (__bf16)hi.z; v[base + 7] = (__bf16)hi.w;
}

// ---------------------------------------------------------------------------
// Projection GEMM: Y[M,N] = X[M,K] * W[N,K]^T (+ bias). N,K compile-time so
// stores/loads fold into immediate offsets. One wave = one 16x16 tile.
// A-frag: lane row M=lane%16; K pack: lane<16 -> {0..7,16..23}, lane>=16 -> +8.
// B-frag: lane column N=lane%16 with identical K packing.
// ---------------------------------------------------------------------------
template <int N, int K>
__global__ __launch_bounds__(256) void proj_gemm_bf16(
    const float* __restrict__ X, const float* __restrict__ W,
    const float* __restrict__ bias, float* __restrict__ Y, int M, int useBias)
{
  const int wave = threadIdx.x >> 5;
  const int lane = threadIdx.x & 31;
  const int gw = blockIdx.x * 8 + wave;
  constexpr int ntn = N >> 4;
  const int total = (M >> 4) * ntn;
  if (gw >= total) return;
  const int tm = gw / ntn;
  const int tn = gw % ntn;

  const int row   = lane & 15;
  const int kbase = (lane < 16) ? 0 : 8;
  const float4* __restrict__ xr4 =
      (const float4*)(X + (size_t)(tm * 16 + row) * K + kbase);
  const float4* __restrict__ wr4 =
      (const float4*)(W + (size_t)(tn * 16 + row) * K + kbase);

  v8f c = {};
#pragma unroll
  for (int k0 = 0; k0 < K; k0 += 32) {
    const int q4 = k0 >> 2;
    v16bf a, b;
    pack8(a, 0, xr4[q4 + 0], xr4[q4 + 1]);  // K = k0+kbase + 0..7
    pack8(a, 8, xr4[q4 + 4], xr4[q4 + 5]);  // K = k0+kbase + 16..23
    pack8(b, 0, wr4[q4 + 0], wr4[q4 + 1]);
    pack8(b, 8, wr4[q4 + 4], wr4[q4 + 5]);
    c = WMMA_BF16(a, b, c);
  }

  // C/D layout: VGPR r -> M=r (lanes 0-15) / M=r+8 (lanes 16-31), N=lane%16
  const int mofs = (lane >> 4) * 8;
  const int gn   = tn * 16 + (lane & 15);
  const float bv = useBias ? bias[gn] : 0.0f;
  float* __restrict__ yp = Y + (size_t)(tm * 16 + mofs) * N + gn;
#pragma unroll
  for (int r = 0; r < 8; ++r) yp[r * N] = c[r] + bv;
}

// ---------------------------------------------------------------------------
// Fused attention: one workgroup (256 thr = 8 waves) per (batch, 16-q tile).
// LDS: s[16][2048] (128KB) + pq[16][128] + wa[128] + 2x feature slab (64KB).
// ---------------------------------------------------------------------------
__global__ __launch_bounds__(256) void fused_attn(
    const float* __restrict__ feature,  // [B,F,D]
    const float* __restrict__ pf,       // [B,F,A]
    const float* __restrict__ pq,       // [B,Q,A] (bias included)
    const float* __restrict__ wa,       // [A]
    const int*   __restrict__ mask,     // [B,F]
    float*       __restrict__ out)      // [B,Q,D]
{
  extern __shared__ float lds[];
  float* s   = lds;                  // [16][kF]
  float* pqs = s + 16 * kF;          // [16][kA]
  float* was = pqs + 16 * kA;        // [kA]
  float* fb0 = was + kA;             // [32][kD] slab buffer 0
  float* fb1 = fb0 + 32 * kD;        // [32][kD] slab buffer 1

  const int tid  = threadIdx.x;
  const int wave = tid >> 5;
  const int lane = tid & 31;
  const int b  = blockIdx.x >> 2;    // Q/16 == 4 query tiles per batch
  const int q0 = (blockIdx.x & 3) * 16;

  for (int i = tid; i < 16 * kA; i += 256)
    pqs[i] = pq[((size_t)b * kQ + q0 + (i >> 7)) * kA + (i & (kA - 1))];
  if (tid < kA) was[tid] = wa[tid];
  __syncthreads();

  // ---- Pass 1: masked scores s[q][f] = sum_a wa[a]*tanh(pf[f][a]+pq[q][a])
  for (int f = tid; f < kF; f += 256) {
    const int mk = mask[b * kF + f];
    float acc[16];
#pragma unroll
    for (int q = 0; q < 16; ++q) acc[q] = 0.0f;
    if (mk) {
      const float4* __restrict__ pr4 =
          (const float4*)(pf + ((size_t)b * kF + f) * kA);
      for (int a0 = 0; a0 < kA; a0 += 8) {
        const float4 p0 = pr4[a0 / 4], p1 = pr4[a0 / 4 + 1];
        const float pv[8] = {p0.x, p0.y, p0.z, p0.w, p1.x, p1.y, p1.z, p1.w};
#pragma unroll
        for (int q = 0; q < 16; ++q) {
          float aq = 0.0f;
#pragma unroll
          for (int j = 0; j < 8; ++j)
            aq += was[a0 + j] * fast_tanh(pv[j] + pqs[q * kA + a0 + j]);
          acc[q] += aq;
        }
      }
    }
#pragma unroll
    for (int q = 0; q < 16; ++q)
      s[q * kF + f] = mk ? acc[q] : -1.0e9f;
  }
  __syncthreads();

  // ---- Pass 2: softmax over F per row (wave32 shuffle reductions) ----------
  for (int q = wave * 2; q < wave * 2 + 2; ++q) {
    float m = -3.0e38f;
    for (int f = lane; f < kF; f += 32) m = fmaxf(m, s[q * kF + f]);
#pragma unroll
    for (int o = 16; o >= 1; o >>= 1) m = fmaxf(m, __shfl_xor(m, o, 32));
    float sum = 0.0f;
    for (int f = lane; f < kF; f += 32) {
      const float e = __expf(s[q * kF + f] - m);
      s[q * kF + f] = e;
      sum += e;
    }
#pragma unroll
    for (int o = 16; o >= 1; o >>= 1) sum += __shfl_xor(sum, o, 32);
    const float inv = __builtin_amdgcn_rcpf(sum);
    for (int f = lane; f < kF; f += 32) s[q * kF + f] *= inv;
  }
  __syncthreads();

  // ---- Pass 3: out[16][256] = alpha[16][2048] @ feature[b] (WMMA bf16) -----
  const int n     = lane & 15;
  const int col0  = wave * 16 + n;          // wave owns column tiles {w, w+8}
  const int col1  = (wave + 8) * 16 + n;
  const int arow  = lane & 15;
  const int kbase = (lane < 16) ? 0 : 8;
  v8f c0 = {}, c1 = {};

#if HAVE_ASYNC_LDS
  const float* fbase = feature + (size_t)b * kF * kD;
  async_slab(fbase, fb0, tid);
  for (int ks = 0; ks < kF / 32; ++ks) {
    float* cur = (ks & 1) ? fb1 : fb0;
    float* nxt = (ks & 1) ? fb0 : fb1;
    const bool more = (ks + 1) < (kF / 32);
    if (more) {
      async_slab(fbase + (size_t)(ks + 1) * 32 * kD, nxt, tid);
      wait_async<8>();   // my 8 loads for `cur` are done (in-order)
    } else {
      wait_async<0>();
    }
    __syncthreads();     // everyone's slab data visible

    const int k0 = ks * 32;
    v16bf af, b0v, b1v;
#pragma unroll
    for (int e = 0; e < 16; ++e) {
      const int kk = kbase + e + ((e >= 8) ? 8 : 0);   // 0..31 within slab
      af[e]  = (__bf16)s[arow * kF + k0 + kk];
      b0v[e] = (__bf16)cur[kk * kD + col0];
      b1v[e] = (__bf16)cur[kk * kD + col1];
    }
    c0 = WMMA_BF16(af, b0v, c0);
    c1 = WMMA_BF16(af, b1v, c1);
    __syncthreads();     // all waves done reading `cur` before it is reused
  }
#else
  for (int k0 = 0; k0 < kF; k0 += 32) {
    if (k0 + 32 < kF)
      __builtin_prefetch(feature + ((size_t)b * kF + k0 + 32 + kbase) * kD, 0, 1);
    v16bf af, b0v, b1v;
#pragma unroll
    for (int e = 0; e < 16; ++e) {
      const int k = k0 + kbase + e + ((e >= 8) ? 8 : 0);
      af[e] = (__bf16)s[arow * kF + k];
      const float* __restrict__ fr = feature + ((size_t)b * kF + k) * kD;
      b0v[e] = (__bf16)fr[col0];
      b1v[e] = (__bf16)fr[col1];
    }
    c0 = WMMA_BF16(af, b0v, c0);
    c1 = WMMA_BF16(af, b1v, c1);
  }
#endif

  const int mofs = (lane >> 4) * 8;
#pragma unroll
  for (int r = 0; r < 8; ++r) {
    const int gq = q0 + mofs + r;
    float* __restrict__ orow = out + ((size_t)b * kQ + gq) * kD;
    orow[col0] = c0[r];
    orow[col1] = c1[r];
  }
}

// ---------------------------------------------------------------------------
extern "C" void kernel_launch(void* const* d_in, const int* in_sizes, int n_in,
                              void* d_out, int out_size, void* d_ws, size_t ws_size,
                              hipStream_t stream) {
  const float* feature = (const float*)d_in[0];  // [B,F,D]
  const float* query   = (const float*)d_in[1];  // [B,Q,QD]
  const float* Wf      = (const float*)d_in[2];  // [A,D]
  const float* Wq      = (const float*)d_in[3];  // [A,QD]
  const float* bq      = (const float*)d_in[4];  // [A]
  const float* wa      = (const float*)d_in[5];  // [A]
  const int*   mask    = (const int*)d_in[6];    // [B,F]
  float* out = (float*)d_out;

  float* pf  = (float*)d_ws;                     // [B,F,A] = 16 MB
  float* pqv = pf + (size_t)kB * kF * kA;        // [B,Q,A] = 0.5 MB

  {  // pf = feature @ Wf^T : M=32768, N=128, K=256
    const int M = kB * kF;
    const int blocks = ((M / 16) * (kA / 16) + 7) / 8;
    proj_gemm_bf16<kA, kD><<<blocks, 256, 0, stream>>>(feature, Wf, nullptr, pf, M, 0);
  }
  {  // pq = query @ Wq^T + bq : M=1024, N=128, K=256
    const int M = kB * kQ;
    const int blocks = ((M / 16) * (kA / 16) + 7) / 8;
    proj_gemm_bf16<kA, kD><<<blocks, 256, 0, stream>>>(query, Wq, bq, pqv, M, 1);
  }
  {  // fused scores + softmax + weighted sum (~200KB LDS, ok on 320KB WGP)
    const size_t shmem =
        (size_t)(16 * kF + 16 * kA + kA + 2 * 32 * kD) * sizeof(float);
    fused_attn<<<kB * (kQ / 16), 256, shmem, stream>>>(feature, pf, pqv, wa, mask, out);
  }
}
